// SmallDecoder_9414568312866
// MI455X (gfx1250) — matrix operand order repaired and checked by reading, not compiled, not run on hardware
//
#include <hip/hip_runtime.h>
#include <hip/hip_bf16.h>

// ---------------------------------------------------------------------------
// Problem constants (from reference)
// ---------------------------------------------------------------------------
#define BATCH   4
#define SEQ     4096
#define DIM     1024          // D
#define NHEAD   16            // H
#define KDIM    64            // K
#define DVEC    1024          // DV
#define FINAL   1024          // H*K
#define ROWS    (BATCH * SEQ) // 16384
#define CHUNK   128
#define NCHUNK  (SEQ / CHUNK) // 32

typedef __attribute__((ext_vector_type(16))) __bf16 v16bf;
typedef __attribute__((ext_vector_type(8)))  __bf16 v8bf;
typedef __attribute__((ext_vector_type(8)))  float  v8f;

// ---------------------------------------------------------------------------
// f32 -> bf16 elementwise convert (for activations)
// ---------------------------------------------------------------------------
__global__ __launch_bounds__(256)
void f32_to_bf16_kernel(const float* __restrict__ in, __bf16* __restrict__ out, int n) {
    int idx = blockIdx.x * 256 + threadIdx.x;
    if (idx < n) out[idx] = (__bf16)in[idx];
}

// ---------------------------------------------------------------------------
// f32 [K][N] row-major -> bf16 [N][K] (transposed, N-major) for weights
// ---------------------------------------------------------------------------
__global__ __launch_bounds__(256)
void transpose_bf16_kernel(const float* __restrict__ in, __bf16* __restrict__ out,
                           int K, int N) {
    long idx = (long)blockIdx.x * 256 + threadIdx.x;
    if (idx < (long)K * N) {
        int k = (int)(idx / N);
        int n = (int)(idx % N);
        out[(long)n * K + k] = (__bf16)in[idx];
    }
}

// ---------------------------------------------------------------------------
// att_raw[row][h] = inputs[row][:] . attW[:][h]   (D x 16, tiny GEMM in f32)
// one block (128 threads) per row
// ---------------------------------------------------------------------------
__global__ __launch_bounds__(128)
void att_raw_kernel(const float* __restrict__ x, const float* __restrict__ W,
                    float* __restrict__ attraw) {
    const int row = blockIdx.x;
    const int t = threadIdx.x;
    float p[NHEAD];
#pragma unroll
    for (int h = 0; h < NHEAD; ++h) p[h] = 0.f;
    const float* xr = x + (size_t)row * DIM;
    for (int d = t; d < DIM; d += 128) {
        float xv = xr[d];
        const float* wr = W + d * NHEAD;
#pragma unroll
        for (int h = 0; h < NHEAD; ++h) p[h] += xv * wr[h];
    }
    __shared__ float red[NHEAD * 128];
#pragma unroll
    for (int h = 0; h < NHEAD; ++h) red[h * 128 + t] = p[h];
    __syncthreads();
    for (int off = 64; off > 0; off >>= 1) {
        if (t < off) {
#pragma unroll
            for (int h = 0; h < NHEAD; ++h) red[h * 128 + t] += red[h * 128 + t + off];
        }
        __syncthreads();
    }
    if (t < NHEAD) attraw[(size_t)row * NHEAD + t] = red[t * 128];
}

// ---------------------------------------------------------------------------
// attmax[b][h] = max_s attraw[b,s,h]   (one block per (b,h))
// ---------------------------------------------------------------------------
__global__ __launch_bounds__(256)
void att_max_kernel(const float* __restrict__ attraw, float* __restrict__ attmax) {
    const int bh = blockIdx.x;          // b*16 + h
    const int b = bh >> 4, h = bh & 15;
    const int t = threadIdx.x;
    float m = -3.4e38f;
    for (int s = t; s < SEQ; s += 256)
        m = fmaxf(m, attraw[((size_t)b * SEQ + s) * NHEAD + h]);
    __shared__ float red[256];
    red[t] = m;
    __syncthreads();
    for (int off = 128; off > 0; off >>= 1) {
        if (t < off) red[t] = fmaxf(red[t], red[t + off]);
        __syncthreads();
    }
    if (t == 0) attmax[bh] = red[0];
}

// ---------------------------------------------------------------------------
// att[idx] = exp((attraw - max) * temperature[h])
// ---------------------------------------------------------------------------
__global__ __launch_bounds__(256)
void att_exp_kernel(const float* __restrict__ attraw, const float* __restrict__ attmax,
                    const float* __restrict__ temp, float* __restrict__ att) {
    int idx = blockIdx.x * 256 + threadIdx.x;       // < ROWS*16
    int h = idx & 15;
    int b = idx >> 16;                              // idx / (SEQ*16)
    att[idx] = __expf((attraw[idx] - attmax[b * NHEAD + h]) * temp[h]);
}

// ---------------------------------------------------------------------------
// bf16 WMMA GEMM:  out[ROWS x 1024] = A[ROWS x Ktot] * Bt^T  (Bt is [1024][Ktot])
// A is split (concat) at k==1024 between A0/A1 (both pitch 1024).
// Block tile 128x128, 8 waves (wave32), wave tile 32x64 -> v_wmma_f32_16x16x32_bf16
// EPI: 0 = plain f32 out
//      1 = + rowVec[b][col], bf16 out          (op = attn_out @ operator + vector)
//      2 = + colBias, relu, bf16 out           (ff1)
//      3 = + colBias + residual[row][col], f32 (ff2 + residual)
// ---------------------------------------------------------------------------
template<int EPI>
__global__ __launch_bounds__(256)
void gemm_bf16_kernel(const __bf16* __restrict__ A0, const __bf16* __restrict__ A1,
                      const __bf16* __restrict__ Bt, int Ktot,
                      float* __restrict__ outF, __bf16* __restrict__ outH,
                      const float* __restrict__ colBias,
                      const float* __restrict__ rowVec,
                      const float* __restrict__ residual) {
    const int tid  = threadIdx.x;
    const int lane = tid & 31;
    const int l    = lane & 15;     // row/col within 16
    const int hf   = lane >> 4;     // half-wave selector
    const int w    = tid >> 5;      // wave id 0..7
    const int wm   = w & 3;         // 4 waves along M
    const int wn   = w >> 2;        // 2 waves along N
    const int rowBase = blockIdx.y * 128 + wm * 32;
    const int colBase = blockIdx.x * 128 + wn * 64;

    v8f acc[2][4];
#pragma unroll
    for (int mi = 0; mi < 2; ++mi)
#pragma unroll
        for (int ni = 0; ni < 4; ++ni)
#pragma unroll
            for (int r = 0; r < 8; ++r) acc[mi][ni][r] = 0.f;

    for (int k0 = 0; k0 < Ktot; k0 += 32) {
        const __bf16* Ap;
        int kk;
        if (k0 < 1024) { Ap = A0; kk = k0; }
        else           { Ap = A1; kk = k0 - 1024; }

        // A fragments: 16-bit A 16x32 layout — lane l holds row m,
        // elements 0..7 = K (hf*8 .. hf*8+7), elements 8..15 = K (16+hf*8 ..)
        v16bf afrag[2];
#pragma unroll
        for (int mi = 0; mi < 2; ++mi) {
            const __bf16* p = Ap + (size_t)(rowBase + mi * 16 + l) * 1024 + kk;
            v8bf lo = *(const v8bf*)(p + hf * 8);
            v8bf hi = *(const v8bf*)(p + 16 + hf * 8);
            afrag[mi] = __builtin_shufflevector(lo, hi,
                0, 1, 2, 3, 4, 5, 6, 7, 8, 9, 10, 11, 12, 13, 14, 15);
        }

        // B fragments: B 32x16 layout — lane l holds col n,
        // elements 0..7 = K (hf*8 .. hf*8+7), elements 8..15 = K (16+hf*8 ..); Bt is N-major so contiguous.
#pragma unroll
        for (int ni = 0; ni < 4; ++ni) {
            const __bf16* q = Bt + (size_t)(colBase + ni * 16 + l) * Ktot + k0 + hf * 8;
            v8bf lo = *(const v8bf*)(q);
            v8bf hi = *(const v8bf*)(q + 16);
            v16bf bfrag = __builtin_shufflevector(lo, hi,
                0, 1, 2, 3, 4, 5, 6, 7, 8, 9, 10, 11, 12, 13, 14, 15);
#pragma unroll
            for (int mi = 0; mi < 2; ++mi) {
                acc[mi][ni] = __builtin_amdgcn_wmma_f32_16x16x32_bf16(
                    false, afrag[mi], false, bfrag, (short)0, acc[mi][ni],
                    false, false);
            }
        }
    }

    // Epilogue: C frag layout — VGPR r holds M = r + 8*hf, N = l
#pragma unroll
    for (int mi = 0; mi < 2; ++mi) {
#pragma unroll
        for (int ni = 0; ni < 4; ++ni) {
#pragma unroll
            for (int r = 0; r < 8; ++r) {
                const int row = rowBase + mi * 16 + r + 8 * hf;
                const int col = colBase + ni * 16 + l;
                float v = acc[mi][ni][r];
                if (EPI == 1) v += rowVec[(row >> 12) * DVEC + col];   // b = row/SEQ
                if (EPI == 2) { v += colBias[col]; v = fmaxf(v, 0.f); }
                if (EPI == 3) { v += colBias[col] + residual[(size_t)row * 1024 + col]; }
                if (EPI == 0 || EPI == 3)
                    outF[(size_t)row * 1024 + col] = v;
                else
                    outH[(size_t)row * 1024 + col] = (__bf16)v;
            }
        }
    }
}

// ---------------------------------------------------------------------------
// Causal prefix-sum over S, chunked. Pass 1: per-chunk column sums.
// grid: b(4) * chunk(32) * jblk(4); 256 threads, 1 column each.
// ---------------------------------------------------------------------------
__global__ __launch_bounds__(256)
void chunk_sum_kernel(const float* __restrict__ v, float* __restrict__ chunk) {
    const int bx = blockIdx.x;
    const int jblk = bx & 3, c = (bx >> 2) & 31, b = bx >> 7;
    const int j = jblk * 256 + threadIdx.x;
    const float* vp = v + ((size_t)b * SEQ + (size_t)c * CHUNK) * 1024 + j;
    float s = 0.f;
    for (int i = 0; i < CHUNK; ++i) s += vp[(size_t)i * 1024];
    chunk[((size_t)b * NCHUNK + c) * 1024 + j] = s;
}

// Pass 2: exclusive scan of the 32 chunk sums per (b, column), in place.
__global__ __launch_bounds__(256)
void chunk_scan_kernel(float* __restrict__ chunk) {
    const int bx = blockIdx.x;              // b(4) * jblk(4)
    const int b = bx >> 2;
    const int j = (bx & 3) * 256 + threadIdx.x;
    float run = 0.f;
    for (int c = 0; c < NCHUNK; ++c) {
        size_t id = ((size_t)b * NCHUNK + c) * 1024 + j;
        float x = chunk[id];
        chunk[id] = run;
        run += x;
    }
}

// Pass 3: rescan within chunk, apply att normalization, emit bf16 attn_out.
// attn_out[b,s,j] = att[b,s,h] * cum / (att[b,s,h]*(s+1) + 1e-30), h = j/64
__global__ __launch_bounds__(256)
void scan_apply_kernel(const float* __restrict__ v, const float* __restrict__ chunk,
                       const float* __restrict__ att, __bf16* __restrict__ attnout) {
    const int bx = blockIdx.x;
    const int jblk = bx & 3, c = (bx >> 2) & 31, b = bx >> 7;
    const int j = jblk * 256 + threadIdx.x;
    const int h = j >> 6;
    float run = chunk[((size_t)b * NCHUNK + c) * 1024 + j];
    for (int i = 0; i < CHUNK; ++i) {
        const int s = c * CHUNK + i;
        const size_t row = (size_t)b * SEQ + s;
        run += v[row * 1024 + j];
        const float a = att[row * NHEAD + h];
        const float o = (a * run) / (a * (float)(s + 1) + 1e-30f);
        attnout[row * 1024 + j] = (__bf16)o;
    }
}

// ---------------------------------------------------------------------------
// LayerNorm over last axis (1024), eps=1e-6. One block per row.
// ---------------------------------------------------------------------------
__global__ __launch_bounds__(256)
void layernorm_kernel(const float* __restrict__ res, const float* __restrict__ gamma,
                      const float* __restrict__ beta, float* __restrict__ out) {
    const int row = blockIdx.x;
    const int t = threadIdx.x;
    const float* r = res + (size_t)row * 1024;
    float x0 = r[t], x1 = r[t + 256], x2 = r[t + 512], x3 = r[t + 768];
    __shared__ float red[256];
    __shared__ float s_mu, s_rstd;
    red[t] = x0 + x1 + x2 + x3;
    __syncthreads();
    for (int off = 128; off > 0; off >>= 1) {
        if (t < off) red[t] += red[t + off];
        __syncthreads();
    }
    if (t == 0) s_mu = red[0] * (1.f / 1024.f);
    __syncthreads();
    const float mu = s_mu;
    const float d0 = x0 - mu, d1 = x1 - mu, d2 = x2 - mu, d3 = x3 - mu;
    red[t] = d0 * d0 + d1 * d1 + d2 * d2 + d3 * d3;
    __syncthreads();
    for (int off = 128; off > 0; off >>= 1) {
        if (t < off) red[t] += red[t + off];
        __syncthreads();
    }
    if (t == 0) s_rstd = rsqrtf(red[0] * (1.f / 1024.f) + 1e-6f);
    __syncthreads();
    const float rs = s_rstd;
    float* o = out + (size_t)row * 1024;
    o[t      ] = d0 * rs * gamma[t      ] + beta[t      ];
    o[t + 256] = d1 * rs * gamma[t + 256] + beta[t + 256];
    o[t + 512] = d2 * rs * gamma[t + 512] + beta[t + 512];
    o[t + 768] = d3 * rs * gamma[t + 768] + beta[t + 768];
}

// ---------------------------------------------------------------------------
// Host launcher
// ---------------------------------------------------------------------------
extern "C" void kernel_launch(void* const* d_in, const int* in_sizes, int n_in,
                              void* d_out, int out_size, void* d_ws, size_t ws_size,
                              hipStream_t stream) {
    (void)in_sizes; (void)n_in; (void)out_size; (void)ws_size;
    const float* inputs   = (const float*)d_in[0];   // (B,S,D)
    const float* vector   = (const float*)d_in[1];   // (B,DV)
    const float* attW     = (const float*)d_in[2];   // (D,H)
    const float* temp     = (const float*)d_in[3];   // (H)
    const float* values   = (const float*)d_in[4];   // (D, H*K) row-major
    const float* oper     = (const float*)d_in[5];   // (FINAL, DV)
    const float* ff1      = (const float*)d_in[6];   // (2048, 1024)
    const float* ff1_bias = (const float*)d_in[7];   // (1024)
    const float* ff2      = (const float*)d_in[8];   // (1024, 1024)
    const float* ff2_bias = (const float*)d_in[9];   // (1024)
    const float* ln_gamma = (const float*)d_in[10];
    const float* ln_beta  = (const float*)d_in[11];
    float* out = (float*)d_out;

    // workspace carve-up (256B aligned)
    char* w = (char*)d_ws;
    size_t off = 0;
    auto take = [&](size_t bytes) { void* p = w + off; off += (bytes + 255) & ~(size_t)255; return p; };
    __bf16* inbf    = (__bf16*)take((size_t)ROWS * 1024 * 2);  // inputs bf16
    __bf16* valT    = (__bf16*)take((size_t)1024 * 1024 * 2);  // values^T
    __bf16* opT     = (__bf16*)take((size_t)1024 * 1024 * 2);  // operator^T
    __bf16* ff1T    = (__bf16*)take((size_t)1024 * 2048 * 2);  // ff1^T
    __bf16* ff2T    = (__bf16*)take((size_t)1024 * 1024 * 2);  // ff2^T
    float*  attraw  = (float*) take((size_t)ROWS * NHEAD * 4);
    float*  att     = (float*) take((size_t)ROWS * NHEAD * 4);
    float*  attmax  = (float*) take((size_t)BATCH * NHEAD * 4);
    float*  chunk   = (float*) take((size_t)BATCH * NCHUNK * 1024 * 4);
    float*  vbuf    = (float*) take((size_t)ROWS * 1024 * 4);  // v; reused as res2
    __bf16* attnout = (__bf16*)take((size_t)ROWS * 1024 * 2);  // reused as res1
    __bf16* opbf    = (__bf16*)take((size_t)ROWS * 1024 * 2);
    float*  res2 = vbuf;
    __bf16* res1 = attnout;

    // 1. convert activations + weights to bf16 (weights transposed to N-major)
    {
        int n = ROWS * 1024;
        f32_to_bf16_kernel<<<n / 256, 256, 0, stream>>>(inputs, inbf, n);
    }
    transpose_bf16_kernel<<<(1024 * 1024) / 256, 256, 0, stream>>>(values, valT, 1024, 1024);
    transpose_bf16_kernel<<<(1024 * 1024) / 256, 256, 0, stream>>>(oper,   opT,  1024, 1024);
    transpose_bf16_kernel<<<(2048 * 1024) / 256, 256, 0, stream>>>(ff1,    ff1T, 2048, 1024);
    transpose_bf16_kernel<<<(1024 * 1024) / 256, 256, 0, stream>>>(ff2,    ff2T, 1024, 1024);

    // 2. attention weights path (f32, tiny)
    att_raw_kernel<<<ROWS, 128, 0, stream>>>(inputs, attW, attraw);
    att_max_kernel<<<BATCH * NHEAD, 256, 0, stream>>>(attraw, attmax);
    att_exp_kernel<<<(ROWS * NHEAD) / 256, 256, 0, stream>>>(attraw, attmax, temp, att);

    // 3. v = inputs @ values  (bf16 WMMA, f32 out)
    gemm_bf16_kernel<0><<<dim3(8, ROWS / 128), 256, 0, stream>>>(
        inbf, nullptr, valT, 1024, vbuf, nullptr, nullptr, nullptr, nullptr);

    // 4. causal prefix-mean with att normalization -> attn_out (bf16)
    chunk_sum_kernel<<<BATCH * NCHUNK * 4, 256, 0, stream>>>(vbuf, chunk);
    chunk_scan_kernel<<<BATCH * 4, 256, 0, stream>>>(chunk);
    scan_apply_kernel<<<BATCH * NCHUNK * 4, 256, 0, stream>>>(vbuf, chunk, att, attnout);

    // 5. op = attn_out @ operator + vector[b]  (bf16 out)
    gemm_bf16_kernel<1><<<dim3(8, ROWS / 128), 256, 0, stream>>>(
        attnout, nullptr, opT, 1024, nullptr, opbf, nullptr, vector, nullptr);

    // 6. res1 = relu([inputs | op] @ ff1 + b1)  (K=2048 concat, bf16 out)
    gemm_bf16_kernel<2><<<dim3(8, ROWS / 128), 256, 0, stream>>>(
        inbf, opbf, ff1T, 2048, nullptr, res1, ff1_bias, nullptr, nullptr);

    // 7. res2 = res1 @ ff2 + b2 + inputs  (f32 out)
    gemm_bf16_kernel<3><<<dim3(8, ROWS / 128), 256, 0, stream>>>(
        res1, nullptr, ff2T, 1024, res2, nullptr, ff2_bias, nullptr, inputs);

    // 8. LayerNorm -> d_out
    layernorm_kernel<<<ROWS, 256, 0, stream>>>(res2, ln_gamma, ln_beta, out);
}
